// HSTUAttention_20160576487489
// MI455X (gfx1250) — compile-verified, matrix operand order checked by reading, not run on hardware
//
#include <hip/hip_runtime.h>
#include <cmath>

// ---------------------------------------------------------------------------
// Problem constants (from reference): B=2, S=2048, D=2048, H=16, HD=128
// ---------------------------------------------------------------------------
constexpr int Bv  = 2;
constexpr int Sv  = 2048;
constexpr int Dv  = 2048;
constexpr int Hv  = 16;
constexpr int HDv = 128;   // head dim
constexpr float EPSv = 1e-5f;

typedef __attribute__((ext_vector_type(16))) __bf16 v16bf;
typedef __attribute__((ext_vector_type(8)))  __bf16 v8bf;
typedef __attribute__((ext_vector_type(8)))  float  v8f;
typedef __attribute__((__vector_size__(16))) int    v4i;   // b128 payload type

#define AS1 __attribute__((address_space(1)))
#define AS3 __attribute__((address_space(3)))

#if __has_builtin(__builtin_amdgcn_global_load_async_to_lds_b128) && \
    __has_builtin(__builtin_amdgcn_s_wait_asynccnt)
#define HAVE_ASYNC_LDS 1
#else
#define HAVE_ASYNC_LDS 0
#endif

__device__ __forceinline__ v8f wmma_bf16(v16bf a, v16bf b, v8f c) {
  // D = A(16x32 bf16) * B(32x16 bf16) + C(16x16 f32)
  return __builtin_amdgcn_wmma_f32_16x16x32_bf16(
      /*neg_a=*/false, a, /*neg_b=*/false, b,
      /*c_mod=*/(short)0, c, /*reuse_a=*/false, /*reuse_b=*/false);
}

__device__ __forceinline__ v8f vzero8() {
  v8f z;
#pragma unroll
  for (int i = 0; i < 8; ++i) z[i] = 0.0f;
  return z;
}

__device__ __forceinline__ v16bf join8(v8bf lo, v8bf hi) {
  return __builtin_shufflevector(lo, hi, 0,1,2,3,4,5,6,7,8,9,10,11,12,13,14,15);
}

// Load one 16-element WMMA operand fragment whose per-lane data is contiguous
// along K: elements 0..7 at base+half*8, elements 8..15 at base+16+half*8.
__device__ __forceinline__ v16bf frag_contig(const __bf16* base, int half) {
  return join8(*(const v8bf*)(base + half * 8),
               *(const v8bf*)(base + 16 + half * 8));
}

__device__ __forceinline__ float silu_f(float x) {
  return x * (1.0f / (1.0f + __expf(-x)));
}

#if HAVE_ASYNC_LDS
__device__ __forceinline__ void async_copy_b128(const __bf16* g, __bf16* l) {
  __builtin_amdgcn_global_load_async_to_lds_b128(
      (AS1 v4i*)(uintptr_t)g,
      (AS3 v4i*)(uint32_t)(uintptr_t)l,
      /*offset=*/0, /*cpol=*/0);
}
#endif

// ---------------------------------------------------------------------------
// Elementwise: h = bf16(silu(x))
// ---------------------------------------------------------------------------
__global__ __launch_bounds__(256) void silu_cast_kernel(const float* __restrict__ x,
                                                        __bf16* __restrict__ h,
                                                        size_t n) {
  size_t i = (size_t)blockIdx.x * 256 + threadIdx.x;
  if (i < n) h[i] = (__bf16)silu_f(x[i]);
}

// ---------------------------------------------------------------------------
// Transpose + cast: wT[N,K] = bf16(w[K,N])  (so WMMA B-operands read
// contiguously along K). Tiled 32x32 through LDS for coalescing.
// grid = (N/32, K/32), block = 256 (32x8 logical)
// ---------------------------------------------------------------------------
__global__ __launch_bounds__(256) void transpose_cast_kernel(const float* __restrict__ w,
                                                             __bf16* __restrict__ wt,
                                                             int K, int N) {
  __shared__ float tile[32][33];
  const int n0 = blockIdx.x * 32;
  const int k0 = blockIdx.y * 32;
  const int tx = threadIdx.x & 31;
  const int ty = threadIdx.x >> 5;   // 0..7
#pragma unroll
  for (int i = 0; i < 32; i += 8)
    tile[ty + i][tx] = w[(size_t)(k0 + ty + i) * N + n0 + tx];
  __syncthreads();
#pragma unroll
  for (int i = 0; i < 32; i += 8)
    wt[(size_t)(n0 + ty + i) * K + k0 + tx] = (__bf16)tile[tx][ty + i];
}

// ---------------------------------------------------------------------------
// Tiled bf16 WMMA GEMM:  C[M,N] = A[M,K] * Bt[N,K]^T
// Both operands staged as 128x32 K-contiguous LDS tiles (identical mapping),
// double-buffered and pipelined on ASYNCcnt: tile k+1 streams into the
// alternate buffer while tile k's WMMAs run; s_wait_asynccnt(4) leaves the
// newest 4 per-thread copies in flight (async ops complete in order).
// Block = 256 threads (8 waves), tile 128x128, K-step 32.
// Wave (wm 0..3, wn 0..1) computes a 32x64 sub-tile = 2x4 WMMA accums.
// OUT_MODE: 0 = f32 [M,N], 1 = bf16 [M,N],
//           2 = bf16 transposed (b,s,h,hd) -> (b,h,hd,s)  (for V)
// ---------------------------------------------------------------------------
template <int OUT_MODE>
__global__ __launch_bounds__(256) void gemm_bf16_kernel(const __bf16* __restrict__ A,
                                                        const __bf16* __restrict__ Bt,
                                                        float* __restrict__ Cf,
                                                        __bf16* __restrict__ Cb,
                                                        int M, int N, int K) {
  __shared__ __bf16 As[2][128 * 32];
  __shared__ __bf16 Bs[2][128 * 32];

  const int lane = threadIdx.x & 31;
  const int wave = threadIdx.x >> 5;
  const int wm   = wave & 3;       // 4 row strips of 32
  const int wn   = wave >> 2;      // 2 col strips of 64
  const int m0   = blockIdx.y * 128;
  const int n0   = blockIdx.x * 128;
  const int r    = lane & 15;
  const int half = lane >> 4;

  v8f acc[2][4];
#pragma unroll
  for (int t = 0; t < 2; ++t)
#pragma unroll
    for (int n = 0; n < 4; ++n) acc[t][n] = vzero8();

  // cooperative tile-load mapping: thread -> (row, 16-col half) of a 128x32 tile
  const int l_row = threadIdx.x >> 1;
  const int l_col = (threadIdx.x & 1) * 16;
  const int l_off = l_row * 32 + l_col;
  const __bf16* gA = A  + (size_t)(m0 + l_row) * K + l_col;
  const __bf16* gB = Bt + (size_t)(n0 + l_row) * K + l_col;

  auto stage_tile = [&](int k0, int sel) {
    __bf16* sA = &As[sel][l_off];
    __bf16* sB = &Bs[sel][l_off];
#if HAVE_ASYNC_LDS
    async_copy_b128(gA + k0,     sA);
    async_copy_b128(gA + k0 + 8, sA + 8);
    async_copy_b128(gB + k0,     sB);
    async_copy_b128(gB + k0 + 8, sB + 8);
#else
    *(v8bf*)(sA)     = *(const v8bf*)(gA + k0);
    *(v8bf*)(sA + 8) = *(const v8bf*)(gA + k0 + 8);
    *(v8bf*)(sB)     = *(const v8bf*)(gB + k0);
    *(v8bf*)(sB + 8) = *(const v8bf*)(gB + k0 + 8);
#endif
  };

  stage_tile(0, 0);
  int sel = 0;
  for (int k0 = 0; k0 < K; k0 += 32, sel ^= 1) {
    if (k0 + 32 < K) {
      stage_tile(k0 + 32, sel ^ 1);
#if HAVE_ASYNC_LDS
      __builtin_amdgcn_s_wait_asynccnt(4);   // tile k landed; k+1 in flight
#endif
    } else {
#if HAVE_ASYNC_LDS
      __builtin_amdgcn_s_wait_asynccnt(0);   // last tile: drain
#endif
    }
    __syncthreads();

    v16bf af[2], bfr[4];
#pragma unroll
    for (int t = 0; t < 2; ++t)
      af[t] = frag_contig(&As[sel][(wm * 32 + t * 16 + r) * 32], half);
#pragma unroll
    for (int n = 0; n < 4; ++n)
      bfr[n] = frag_contig(&Bs[sel][(wn * 64 + n * 16 + r) * 32], half);

#pragma unroll
    for (int t = 0; t < 2; ++t)
#pragma unroll
      for (int n = 0; n < 4; ++n) acc[t][n] = wmma_bf16(af[t], bfr[n], acc[t][n]);
    __syncthreads();   // nobody still reads buf 'sel' when it is re-staged
  }

  // store: C/D layout -> row = 8*half + j, col = r (lane 0-15 / 16-31 halves)
#pragma unroll
  for (int t = 0; t < 2; ++t)
#pragma unroll
    for (int n = 0; n < 4; ++n) {
      const int row0 = m0 + wm * 32 + t * 16 + half * 8;
      const int col  = n0 + wn * 64 + n * 16 + r;
      if (OUT_MODE == 2) {
        // (b,s) = row, (h,hd) = col ; dest layout (b,h,hd,s); 8 consecutive s
        const int bb = row0 / Sv, s0 = row0 % Sv;
        const int hh = col / HDv, hd = col % HDv;
        v8bf pk;
#pragma unroll
        for (int j = 0; j < 8; ++j) pk[j] = (__bf16)acc[t][n][j];
        *(v8bf*)(Cb + (((size_t)bb * Hv + hh) * HDv + hd) * Sv + s0) = pk;
      } else {
#pragma unroll
        for (int j = 0; j < 8; ++j) {
          if (OUT_MODE == 1) Cb[(size_t)(row0 + j) * N + col] = (__bf16)acc[t][n][j];
          else               Cf[(size_t)(row0 + j) * N + col] = acc[t][n][j];
        }
      }
    }
}

// ---------------------------------------------------------------------------
// RoPE (interleaved pairs) fp32 -> bf16, layout (B,S,H,HD)
// ---------------------------------------------------------------------------
__global__ __launch_bounds__(256) void rope_cast_kernel(const float* __restrict__ src,
                                                        const float* __restrict__ fc,
                                                        const float* __restrict__ fs,
                                                        __bf16* __restrict__ dst) {
  size_t idx = (size_t)blockIdx.x * 256 + threadIdx.x;   // over pairs
  size_t total = (size_t)Bv * Sv * Hv * (HDv / 2);
  if (idx >= total) return;
  int    p  = (int)(idx % (HDv / 2));
  size_t t  = idx / (HDv / 2);
  size_t bs = t / Hv;
  int    s  = (int)(bs % Sv);
  float c  = fc[(size_t)s * (HDv / 2) + p];
  float sn = fs[(size_t)s * (HDv / 2) + p];
  size_t base = idx * 2;
  float re = src[base], im = src[base + 1];
  dst[base]     = (__bf16)(re * c - im * sn);
  dst[base + 1] = (__bf16)(re * sn + im * c);
}

// ---------------------------------------------------------------------------
// Causal silu-attention (HSTU): one wave owns a 16-row Q strip.
//   scores = silu(q.k^T / sqrt(HD)) masked causally (no softmax)
//   out    = scores @ v
// q,k layout (B,S,H,HD); v layout (B,H,HD,S) [pre-transposed so P.V reads
// contiguously along keys]. grid = (S/128, H, B), block = 256 (8 waves).
// ---------------------------------------------------------------------------
__global__ __launch_bounds__(256) void attn_kernel(const __bf16* __restrict__ q,
                                                   const __bf16* __restrict__ k,
                                                   const __bf16* __restrict__ vT,
                                                   float* __restrict__ o) {
  __shared__ __bf16 Pbuf[8][16 * 32];   // per-wave score staging (C-layout -> A-layout)

  const int lane  = threadIdx.x & 31;
  const int wave  = threadIdx.x >> 5;
  const int hh    = blockIdx.y;
  const int bb    = blockIdx.z;
  const int qbase = blockIdx.x * 128 + wave * 16;
  const int r     = lane & 15;
  const int half  = lane >> 4;
  const size_t kstride = (size_t)Hv * HDv;      // seq stride in q/k layout
  const float  scale   = 0.08838834764831845f;  // 1/sqrt(128)

  // Q A-fragments for this wave's 16 rows, 4 chunks of 32 head-dims
  const __bf16* qrow = q + (((size_t)bb * Sv + qbase + r) * Hv + hh) * HDv;
  v16bf qa[4];
#pragma unroll
  for (int d = 0; d < 4; ++d) qa[d] = frag_contig(qrow + d * 32, half);

  v8f acc[8];
#pragma unroll
  for (int n = 0; n < 8; ++n) acc[n] = vzero8();

  const int kend = qbase + 16;           // causal frontier (exclusive)
  for (int kb = 0; kb < kend; kb += 32) {
    // ---- scores: two 16x16 tiles over keys [kb,kb+16) and [kb+16,kb+32)
    v8f s0 = vzero8(), s1 = vzero8();
    const __bf16* krow0 = k + (((size_t)bb * Sv + kb + r) * Hv + hh) * HDv;
    const __bf16* krow1 = krow0 + 16 * kstride;
#pragma unroll
    for (int d = 0; d < 4; ++d) {
      s0 = wmma_bf16(qa[d], frag_contig(krow0 + d * 32, half), s0);
      s1 = wmma_bf16(qa[d], frag_contig(krow1 + d * 32, half), s1);
    }

    // ---- scale + silu + causal mask; stage through LDS in A-layout order
#pragma unroll
    for (int j = 0; j < 8; ++j) {
      int qg   = qbase + half * 8 + j;    // global query row for this accum slot
      float x0 = s0[j] * scale;
      float x1 = s1[j] * scale;
      x0 = (kb + r      <= qg) ? silu_f(x0) : 0.0f;
      x1 = (kb + 16 + r <= qg) ? silu_f(x1) : 0.0f;
      Pbuf[wave][(half * 8 + j) * 32 + r]      = (__bf16)x0;
      Pbuf[wave][(half * 8 + j) * 32 + 16 + r] = (__bf16)x1;
    }
    // per-wave LDS is in-order (DScnt) — no cross-wave barrier needed

    // ---- P (16x32) @ V (32x128): A-fragment from LDS, B from (B,H,HD,S) vT
    v16bf pa = frag_contig(&Pbuf[wave][r * 32], half);
#pragma unroll
    for (int n = 0; n < 8; ++n) {
      const __bf16* vcol = vT + (((size_t)bb * Hv + hh) * HDv + n * 16 + r) * Sv + kb;
      acc[n] = wmma_bf16(pa, frag_contig(vcol, half), acc[n]);
    }
  }

  // ---- store 16x128 f32 tile to attention output, layout (B,S,H,HD)
#pragma unroll
  for (int n = 0; n < 8; ++n)
#pragma unroll
    for (int j = 0; j < 8; ++j) {
      int row = qbase + half * 8 + j;
      o[(((size_t)bb * Sv + row) * Hv + hh) * HDv + n * 16 + r] = acc[n][j];
    }
}

// ---------------------------------------------------------------------------
// RMS-norm over D, * norm_w, * u  -> bf16; one block per (b,s) row
// ---------------------------------------------------------------------------
__global__ __launch_bounds__(256) void rmsnorm_mul_kernel(const float* __restrict__ a,
                                                          const float* __restrict__ u,
                                                          const float* __restrict__ nw,
                                                          __bf16* __restrict__ out) {
  __shared__ float red[256];
  const size_t row = blockIdx.x;
  const float* ar = a + row * Dv;
  float ss = 0.0f;
  for (int i = threadIdx.x; i < Dv; i += 256) { float t = ar[i]; ss += t * t; }
  red[threadIdx.x] = ss;
  __syncthreads();
  for (int s = 128; s > 0; s >>= 1) {
    if ((int)threadIdx.x < s) red[threadIdx.x] += red[threadIdx.x + s];
    __syncthreads();
  }
  float rs = rsqrtf(red[0] / (float)Dv + EPSv);
  const float* ur = u + row * Dv;
  __bf16* orow = out + row * Dv;
  for (int i = threadIdx.x; i < Dv; i += 256)
    orow[i] = (__bf16)(ar[i] * rs * nw[i] * ur[i]);
}

// ---------------------------------------------------------------------------
// Host-side orchestration
// ---------------------------------------------------------------------------
extern "C" void kernel_launch(void* const* d_in, const int* in_sizes, int n_in,
                              void* d_out, int out_size, void* d_ws, size_t ws_size,
                              hipStream_t stream) {
  const float* x    = (const float*)d_in[0];
  const float* fcos = (const float*)d_in[1];
  const float* fsin = (const float*)d_in[2];
  const float* wq   = (const float*)d_in[3];
  const float* wk   = (const float*)d_in[4];
  const float* wv   = (const float*)d_in[5];
  const float* wu   = (const float*)d_in[6];
  const float* wo   = (const float*)d_in[7];
  const float* nw   = (const float*)d_in[8];
  float* out = (float*)d_out;

  const size_t MS = (size_t)Bv * Sv;     // 4096 rows
  const size_t ND = (size_t)MS * Dv;     // 8M activations
  const size_t WD = (size_t)Dv * Dv;     // 4M weights

  char* ws = (char*)d_ws;
  auto carve = [&](size_t bytes) { char* p = ws; ws += (bytes + 255) & ~(size_t)255; return p; };

  __bf16* h_bf  = (__bf16*)carve(ND * 2);
  __bf16* wqT   = (__bf16*)carve(WD * 2);
  __bf16* wkT   = (__bf16*)carve(WD * 2);
  __bf16* wvT   = (__bf16*)carve(WD * 2);
  __bf16* wuT   = (__bf16*)carve(WD * 2);
  __bf16* woT   = (__bf16*)carve(WD * 2);
  float*  q_f   = (float*)carve(ND * 4);
  float*  k_f   = (float*)carve(ND * 4);
  float*  u_f   = (float*)carve(ND * 4);
  __bf16* q_bf  = (__bf16*)carve(ND * 2);
  __bf16* k_bf  = (__bf16*)carve(ND * 2);
  __bf16* vT_bf = (__bf16*)carve(ND * 2);
  // q_f / k_f are dead after RoPE: reuse them for attention output + normed acts
  float*  attn_f = q_f;                  // ND f32
  __bf16* n_bf   = (__bf16*)k_f;         // ND bf16 (fits in ND f32 region)

  // 1) h = silu(x) in bf16; weights -> bf16 transposed [N,K]
  silu_cast_kernel<<<(int)((ND + 255) / 256), 256, 0, stream>>>(x, h_bf, ND);
  dim3 tgrid(Dv / 32, Dv / 32);
  transpose_cast_kernel<<<tgrid, 256, 0, stream>>>(wq, wqT, Dv, Dv);
  transpose_cast_kernel<<<tgrid, 256, 0, stream>>>(wk, wkT, Dv, Dv);
  transpose_cast_kernel<<<tgrid, 256, 0, stream>>>(wv, wvT, Dv, Dv);
  transpose_cast_kernel<<<tgrid, 256, 0, stream>>>(wu, wuT, Dv, Dv);
  transpose_cast_kernel<<<tgrid, 256, 0, stream>>>(wo, woT, Dv, Dv);

  // 2) projections (M=4096, N=K=2048), WMMA bf16
  dim3 ggrid(Dv / 128, (int)(MS / 128));
  gemm_bf16_kernel<0><<<ggrid, 256, 0, stream>>>(h_bf, wqT, q_f, nullptr, (int)MS, Dv, Dv);
  gemm_bf16_kernel<0><<<ggrid, 256, 0, stream>>>(h_bf, wkT, k_f, nullptr, (int)MS, Dv, Dv);
  gemm_bf16_kernel<2><<<ggrid, 256, 0, stream>>>(h_bf, wvT, nullptr, vT_bf, (int)MS, Dv, Dv);
  gemm_bf16_kernel<0><<<ggrid, 256, 0, stream>>>(h_bf, wuT, u_f, nullptr, (int)MS, Dv, Dv);

  // 3) RoPE on q,k -> bf16
  size_t pairs = ND / 2;
  rope_cast_kernel<<<(int)((pairs + 255) / 256), 256, 0, stream>>>(q_f, fcos, fsin, q_bf);
  rope_cast_kernel<<<(int)((pairs + 255) / 256), 256, 0, stream>>>(k_f, fcos, fsin, k_bf);

  // 4) causal silu-attention (WMMA bf16, f32 accum)
  attn_kernel<<<dim3(Sv / 128, Hv, Bv), 256, 0, stream>>>(q_bf, k_bf, vT_bf, attn_f);

  // 5) rmsnorm * norm_w * u -> bf16
  rmsnorm_mul_kernel<<<(int)MS, 256, 0, stream>>>(attn_f, u_f, nw, n_bf);

  // 6) final projection -> f32 output
  gemm_bf16_kernel<0><<<ggrid, 256, 0, stream>>>(n_bf, woT, out, nullptr, (int)MS, Dv, Dv);
}
